// MPNNModel_73289321939188
// MI455X (gfx1250) — matrix operand order repaired and checked by reading, not compiled, not run on hardware
//
#include <hip/hip_runtime.h>
#include <hip/hip_bf16.h>
#include <stdint.h>

#define NNODES 50000
#define NEDGES 600000
#define HD 128
#define BN_EPS 1e-3f

typedef __attribute__((ext_vector_type(16))) __bf16 v16bf;
typedef __attribute__((ext_vector_type(8)))  float  v8f;

__device__ __forceinline__ uint16_t f2bf(float x) {
  uint32_t u = __builtin_bit_cast(uint32_t, x);
  u += 0x7FFFu + ((u >> 16) & 1u);   // round-to-nearest-even
  return (uint16_t)(u >> 16);
}

union V16U8 { uint32_t u[8]; v16bf v; };

// ---------------------------------------------------------------------------
// Pack weight matrix B [K=128, N] (optionally the concat [W_top;W_bot] form of
// W[i] in [2H,H]) into a contiguous-k (matching A) v_wmma_f32_16x16x32_bf16 B-operand register
// layout: out[((nt*4 + ks)*256) + lane*8 + j] holds packed {B[k],B[k+1]} for
// k = ks*32 + (lane<16?0:16) + 2j, n = nt*16 + (lane&15).
// ---------------------------------------------------------------------------
__global__ __launch_bounds__(256) void pack_b_kernel(
    const float* __restrict__ W, uint32_t* __restrict__ out,
    int ntiles, int concat) {
  int idx = blockIdx.x * blockDim.x + threadIdx.x;
  int total = ntiles * 4 * 256;
  if (idx >= total) return;
  int j    = idx & 7;
  int lane = (idx >> 3) & 31;
  int ks   = (idx >> 8) & 3;
  int nt   = idx >> 10;
  int col  = lane & 15;
  int kb   = (lane < 16) ? 0 : 16;
  int k    = ks * 32 + kb + 2 * j;
  int n    = nt * 16 + col;
  float lo, hi;
  if (concat && n >= 128) {  // bottom half of W[i]: rows 128..255, cols n-128
    lo = W[(size_t)(128 + k) * 128 + (n - 128)];
    hi = W[(size_t)(129 + k) * 128 + (n - 128)];
  } else {
    lo = W[(size_t)k       * 128 + n];
    hi = W[(size_t)(k + 1) * 128 + n];
  }
  out[idx] = (uint32_t)f2bf(lo) | ((uint32_t)f2bf(hi) << 16);
}

// ---------------------------------------------------------------------------
// GEMM: C[M,N] = A[M,128] (bf16) @ Bpk (pre-packed bf16) [+ bias], f32 accum
// via v_wmma_f32_16x16x32_bf16. Block = 256 threads = 8 waves. Each wave loads
// its 16x128 A slab into registers ONCE (8 x uint4), then computes N/128
// 16x16 output tiles (nt = wave, wave+8, ...), 4 WMMAs each over K=128.
// EMB=true additionally adds bias and emits a bf16 copy of C.
// ---------------------------------------------------------------------------
template <int N, bool EMB>
__global__ __launch_bounds__(256) void gemm_wmma_bf16(
    const uint16_t* __restrict__ A, const uint32_t* __restrict__ Bpk,
    float* __restrict__ C, uint16_t* __restrict__ Cbf,
    const float* __restrict__ bias) {
  const int lane = threadIdx.x & 31;
  const int wave = threadIdx.x >> 5;
  const int m0   = blockIdx.x * 16;
  const int row  = m0 + (lane & 15);
  const uint32_t* arow = (const uint32_t*)(A + (size_t)row * HD); // 64 dwords
  const int pb = (lane < 16) ? 0 : 8;  // K-pair base per half-wave (A layout)

  // A tile for this wave: resident in registers across all N-tiles.
  uint4 areg[8];
#pragma unroll
  for (int ks = 0; ks < 4; ++ks) {
    areg[2 * ks]     = *(const uint4*)(arow + ks * 16 + pb);
    areg[2 * ks + 1] = *(const uint4*)(arow + ks * 16 + 4 + pb);
  }

  const int col = lane & 15;
  const int mb  = (lane < 16) ? 0 : 8;  // C/D layout: VGPR r -> M = mb + r

  constexpr int NT_PER_WAVE = N / 128;  // 1 (emb) or 2 (conv)
#pragma unroll
  for (int t = 0; t < NT_PER_WAVE; ++t) {
    const int nt = t * 8 + wave;
    v8f c = {0.f, 0.f, 0.f, 0.f, 0.f, 0.f, 0.f, 0.f};
#pragma unroll
    for (int ks = 0; ks < 4; ++ks) {
      V16U8 av, bv;
      *(uint4*)(&av.u[0]) = areg[2 * ks];
      *(uint4*)(&av.u[4]) = areg[2 * ks + 1];
      const uint32_t* bb = Bpk + ((size_t)(nt * 4 + ks) * 256) + lane * 8;
      *(uint4*)(&bv.u[0]) = *(const uint4*)(bb);
      *(uint4*)(&bv.u[4]) = *(const uint4*)(bb + 4);
      c = __builtin_amdgcn_wmma_f32_16x16x32_bf16(
          /*neg_a=*/false, av.v, /*neg_b=*/false, bv.v,
          /*c_mod=*/(short)0, c, /*reuse_a=*/false, /*reuse_b=*/false);
    }
    const int n = nt * 16 + col;
    float bias_v = 0.0f;
    if constexpr (EMB) bias_v = bias[n];
#pragma unroll
    for (int r = 0; r < 8; ++r) {
      float v = c[r] + bias_v;
      int m = m0 + mb + r;
      C[(size_t)m * N + n] = v;
      if constexpr (EMB) Cbf[(size_t)m * HD + n] = f2bf(v);
    }
  }
}

__global__ __launch_bounds__(256) void cvt_bf16_kernel(
    const float* __restrict__ x, uint16_t* __restrict__ y, int n) {
  int i = blockIdx.x * blockDim.x + threadIdx.x;
  if (i < n) y[i] = f2bf(x[i]);
}

__global__ __launch_bounds__(256) void zero_u32_kernel(uint32_t* __restrict__ p, int n) {
  int i = blockIdx.x * blockDim.x + threadIdx.x;
  if (i < n) p[i] = 0u;
}

__global__ __launch_bounds__(256) void deg_kernel(
    const int* __restrict__ dst, int* __restrict__ deg, int E) {
  int e = blockIdx.x * blockDim.x + threadIdx.x;
  if (e < E) atomicAdd(&deg[dst[e]], 1);
}

// 32 lanes per edge, float4 gather of p_src[src[e]] (cols 0..127 of P), 4
// coalesced f32 atomic adds into agg[dst[e]] (L2-resident).
__global__ __launch_bounds__(256) void scatter_kernel(
    const float* __restrict__ P, const int* __restrict__ src,
    const int* __restrict__ dst, float* __restrict__ agg, int E) {
  int gid = blockIdx.x * blockDim.x + threadIdx.x;
  int e = gid >> 5;
  if (e >= E) return;
  int f0 = (gid & 31) * 4;
  int s = src[e], d = dst[e];
  const float4 v = *(const float4*)(P + (size_t)s * 256 + f0);
  float* a = agg + (size_t)d * HD + f0;
  __hip_atomic_fetch_add(a + 0, v.x, __ATOMIC_RELAXED, __HIP_MEMORY_SCOPE_AGENT);
  __hip_atomic_fetch_add(a + 1, v.y, __ATOMIC_RELAXED, __HIP_MEMORY_SCOPE_AGENT);
  __hip_atomic_fetch_add(a + 2, v.z, __ATOMIC_RELAXED, __HIP_MEMORY_SCOPE_AGENT);
  __hip_atomic_fetch_add(a + 3, v.w, __ATOMIC_RELAXED, __HIP_MEMORY_SCOPE_AGENT);
}

// Per-node-feature: a = agg + deg*(p_dst + b); h = relu(BN(sigmoid(a)+softplus(h)))
__global__ __launch_bounds__(256) void update_kernel(
    const float* __restrict__ agg, const float* __restrict__ P,
    const int* __restrict__ deg,
    const float* __restrict__ bconv, const float* __restrict__ gamma,
    const float* __restrict__ beta, const float* __restrict__ mean,
    const float* __restrict__ var,
    float* __restrict__ h, uint16_t* __restrict__ hbf, int n) {
  int i = blockIdx.x * blockDim.x + threadIdx.x;
  if (i >= n) return;
  int v = i >> 7, f = i & 127;
  float dg = (float)deg[v];
  float a  = agg[i] + dg * (P[(size_t)v * 256 + 128 + f] + bconv[f]);
  float s  = 1.0f / (1.0f + expf(-a));                       // sigmoid(agg)
  float hv = h[i];
  float sp = fmaxf(hv, 0.0f) + log1pf(expf(-fabsf(hv)));     // softplus(h)
  float y  = s + sp;
  y = gamma[f] * (y - mean[f]) * rsqrtf(var[f] + BN_EPS) + beta[f];
  y = fmaxf(y, 0.0f);                                        // ReLU
  h[i] = y;
  hbf[i] = f2bf(y);
}

extern "C" void kernel_launch(void* const* d_in, const int* in_sizes, int n_in,
                              void* d_out, int out_size, void* d_ws, size_t ws_size,
                              hipStream_t stream) {
  const float* node_feat = (const float*)d_in[0];
  const int*   src       = (const int*)d_in[1];
  const int*   dst       = (const int*)d_in[2];
  const float* W_emb     = (const float*)d_in[3];
  const float* b_emb     = (const float*)d_in[4];
  const float* W         = (const float*)d_in[5];   // [2, 256, 128]
  const float* b         = (const float*)d_in[6];   // [2, 128]
  const float* gamma     = (const float*)d_in[7];
  const float* beta      = (const float*)d_in[8];
  const float* bn_mean   = (const float*)d_in[9];
  const float* bn_var    = (const float*)d_in[10];

  // Workspace carve-out (≈103 MB total)
  char* ws = (char*)d_ws;
  size_t off = 0;
  auto take = [&](size_t bytes) -> char* {
    char* p = ws + off;
    off = (off + bytes + 255) & ~(size_t)255;
    return p;
  };
  uint16_t* hbfA = (uint16_t*)take((size_t)NNODES * HD * 2);   // bf16 h ping
  uint16_t* hbfB = (uint16_t*)take((size_t)NNODES * HD * 2);   // bf16 h pong
  float*    P    = (float*)   take((size_t)NNODES * 256 * 4);  // [p_src | p_dst]
  float*    agg  = (float*)   take((size_t)NNODES * HD * 4);
  int*      deg  = (int*)     take((size_t)NNODES * 4);
  uint32_t* BpkE = (uint32_t*)take((size_t)8  * 4 * 256 * 4);  // packed W_emb
  uint32_t* Bpk0 = (uint32_t*)take((size_t)16 * 4 * 256 * 4);  // packed W[0]
  uint32_t* Bpk1 = (uint32_t*)take((size_t)16 * 4 * 256 * 4);  // packed W[1]
  float* h = (float*)d_out;                                    // fp32 h [N,128]

  // 1) Pack weights into WMMA B layout; convert node_feat to bf16; degrees.
  pack_b_kernel<<<32, 256, 0, stream>>>(W_emb, BpkE, 8, 0);
  pack_b_kernel<<<64, 256, 0, stream>>>(W,             Bpk0, 16, 1);
  pack_b_kernel<<<64, 256, 0, stream>>>(W + 256 * 128, Bpk1, 16, 1);
  cvt_bf16_kernel<<<(NNODES * HD + 255) / 256, 256, 0, stream>>>(node_feat, hbfA, NNODES * HD);
  zero_u32_kernel<<<(NNODES + 255) / 256, 256, 0, stream>>>((uint32_t*)deg, NNODES);
  deg_kernel<<<(NEDGES + 255) / 256, 256, 0, stream>>>(dst, deg, NEDGES);

  // 2) Embedding: h = node_feat @ W_emb + b_emb  (fp32 to d_out, bf16 to hbfB)
  gemm_wmma_bf16<128, true><<<dim3(NNODES / 16), 256, 0, stream>>>(hbfA, BpkE, h, hbfB, b_emb);

  // 3) Two MPNN convolutions
  for (int i = 0; i < 2; ++i) {
    const uint16_t* hin  = (i == 0) ? hbfB : hbfA;
    uint16_t*       hout = (i == 0) ? hbfA : hbfB;
    const uint32_t* Bpk  = (i == 0) ? Bpk0 : Bpk1;
    // P = h @ [W_top | W_bot]   (N = 256); A slab register-resident per wave
    gemm_wmma_bf16<256, false><<<dim3(NNODES / 16), 256, 0, stream>>>(hin, Bpk, P, nullptr, nullptr);
    // agg = scatter-add of p_src over dst
    zero_u32_kernel<<<(NNODES * HD + 255) / 256, 256, 0, stream>>>((uint32_t*)agg, NNODES * HD);
    scatter_kernel<<<(NEDGES * 32 + 255) / 256, 256, 0, stream>>>(P, src, dst, agg, NEDGES);
    // node update: sigmoid/softplus + BN + ReLU; writes fp32 h and bf16 h
    update_kernel<<<(NNODES * HD + 255) / 256, 256, 0, stream>>>(
        agg, P, deg, b + i * HD, gamma + i * HD, beta + i * HD,
        bn_mean + i * HD, bn_var + i * HD, h, hout, NNODES * HD);
  }
}
